// DecoderRNN_65498251264267
// MI455X (gfx1250) — matrix operand order repaired and checked by reading, not compiled, not run on hardware
//
#include <hip/hip_runtime.h>
#include <cstdint>

// Problem dimensions (from reference)
#define T_T   64
#define B_B   256
#define FEAT  924
#define H     512
#define NC    151
#define ED    100
#define KIN   1024   // FEAT + ED
#define SIXH  3072   // 6*H
#define FIVEH 2560   // 5*H
#define NCPAD 160    // NC padded to a multiple of 16

// Pi-GEMM block tiling
#define BLK_M 128
#define BLK_N 64
#define BLK_K 32
#define LDS_LD 40    // padded LDS row stride (elements); 80B keeps 16B alignment, spreads banks

typedef __bf16 bf16;
typedef __attribute__((ext_vector_type(16))) __bf16 v16bf;
typedef __attribute__((ext_vector_type(8)))  __bf16 v8bf;
typedef __attribute__((ext_vector_type(8)))  float  v8f;

__device__ __forceinline__ int laneid() { return threadIdx.x & 31; }

// Low 32 bits of a generic LDS address = wave-relative LDS byte offset
// (flat aperture: addr[63:32] = SHARED_BASE, addr[31:0] = LDS offset).
__device__ __forceinline__ uint32_t lds_off(const void* p) {
  return (uint32_t)(uintptr_t)p;
}

// CDNA5 async DMA: global memory -> LDS, 16 bytes per lane, tracked by ASYNCcnt.
__device__ __forceinline__ void async_copy_b128(uint32_t lds_byte, const bf16* g) {
  uint64_t ga = (uint64_t)(uintptr_t)g;
  asm volatile("global_load_async_to_lds_b128 %0, %1, off"
               :: "v"(lds_byte), "v"(ga) : "memory");
}

__device__ __forceinline__ void wait_async0() {
  asm volatile("s_wait_asynccnt 0x0" ::: "memory");
}

// A-fragment (16x32 bf16, M x K) from a row-major [rows x ld] bf16 matrix.
// ISA layout: lane m (0-15): K = {0..7, 16..23}; lane m+16: K = {8..15, 24..31}.
__device__ __forceinline__ v16bf load_fragA(const bf16* __restrict__ base,
                                            int row0, int ld, int k0) {
  const int lane = laneid();
  const int m    = lane & 15;
  const int hi8  = (lane >> 4) << 3;
  const bf16* p  = base + (size_t)(row0 + m) * ld + (k0 + hi8);
  v8bf lo = *(const v8bf*)(p);
  v8bf hi = *(const v8bf*)(p + 16);
  v16bf out;
#pragma unroll
  for (int i = 0; i < 8; ++i) { out[i] = lo[i]; out[i + 8] = hi[i]; }
  return out;
}

// B-fragment (32x16 bf16, K x N) where the logical B is W^T of a row-major
// weight matrix W [N x K]: lane n (0-15) holds column n, K = {0..7, 16..23};
// lane n+16 holds K = {8..15, 24..31} (same K order as the A fragment).
__device__ __forceinline__ v16bf load_fragB(const bf16* __restrict__ wrow,
                                            int row0, int ld, int k0) {
  const int lane = laneid();
  const int n    = lane & 15;
  const int koff = (lane >> 4) << 3;
  const bf16* p  = wrow + (size_t)(row0 + n) * ld + (k0 + koff);
  v8bf lo = *(const v8bf*)(p);
  v8bf hi = *(const v8bf*)(p + 16);
  v16bf out;
#pragma unroll
  for (int i = 0; i < 8; ++i) { out[i] = lo[i]; out[i + 8] = hi[i]; }
  return out;
}

__device__ __forceinline__ float sigm(float x) { return 1.0f / (1.0f + __expf(-x)); }

// ---------------------------------------------------------------- prep kernels
__global__ void cvt_f32_bf16(const float* __restrict__ src, bf16* __restrict__ dst, int n) {
  int i = blockIdx.x * blockDim.x + threadIdx.x;
  if (i < n) dst[i] = (bf16)src[i];
}

__global__ void cvt_wo_kernel(const float* __restrict__ wo, bf16* __restrict__ dst) {
  int i = blockIdx.x * blockDim.x + threadIdx.x;
  if (i >= NCPAD * H) return;
  int r = i / H;
  int c = i - r * H;
  dst[i] = (r < NC) ? (bf16)wo[(size_t)r * H + c] : (bf16)0.0f;
}

// TI[t*B+b][0:924] = x[t][b][:]; TI[t*B+b][924:1024] = embed[labels[t][b]][:]
__global__ void build_ti_kernel(const float* __restrict__ x, const int* __restrict__ labels,
                                const float* __restrict__ embed, bf16* __restrict__ ti) {
  size_t i = (size_t)blockIdx.x * blockDim.x + threadIdx.x;
  if (i >= (size_t)T_T * B_B * KIN) return;
  int    col = (int)(i & (KIN - 1));
  size_t row = i >> 10;          // t*B + b
  float v;
  if (col < FEAT) {
    v = x[row * FEAT + col];
  } else {
    int lab = labels[row];
    v = embed[(size_t)lab * ED + (col - FEAT)];
  }
  ti[i] = (bf16)v;
}

__global__ void init_state_kernel(bf16* __restrict__ h0, float* __restrict__ c) {
  int i = blockIdx.x * blockDim.x + threadIdx.x;
  if (i < B_B * H) { h0[i] = (bf16)0.0f; c[i] = 0.0f; }
}

// ------------------------------------------------- big input GEMM (all T at once)
// Pi[T*B, 6H] = TI[T*B, 1024] @ Wi[6H, 1024]^T + bi
// Block-cooperative: 256 threads / 8 waves, 128x64 block tile, K staged in 32-wide
// slabs through double-buffered LDS via global_load_async_to_lds_b128 (ASYNCcnt).
__global__ void __launch_bounds__(256) gemm_pi_kernel(const bf16* __restrict__ ti,
                                                      const bf16* __restrict__ wi,
                                                      const float* __restrict__ bi,
                                                      float* __restrict__ pi) {
  __shared__ bf16 sA[2][BLK_M * LDS_LD];  // 2 x 10 KB
  __shared__ bf16 sB[2][BLK_N * LDS_LD];  // 2 x  5 KB

  const int tid  = threadIdx.x;
  const int wv   = tid >> 5;
  const int mB   = blockIdx.x / (SIXH / BLK_N);
  const int nB   = blockIdx.x % (SIXH / BLK_N);
  const int rowG = mB * BLK_M;
  const int n0   = nB * BLK_N;

  // Issue async DMA of one K-slab (A: 128x32, B: 64x32) into buffer `buf`.
  auto stage = [&](int buf, int k) {
    // A tile: 512 16B chunks, 2 per thread
#pragma unroll
    for (int it = 0; it < 2; ++it) {
      const int chunk = tid + it * 256;
      const int r = chunk >> 2, cch = chunk & 3;
      const bf16* g = ti + (size_t)(rowG + r) * KIN + k + cch * 8;
      async_copy_b128(lds_off(&sA[buf][r * LDS_LD + cch * 8]), g);
    }
    // B tile: 256 16B chunks, 1 per thread
    {
      const int r = tid >> 2, cch = tid & 3;
      const bf16* g = wi + (size_t)(n0 + r) * KIN + k + cch * 8;
      async_copy_b128(lds_off(&sB[buf][r * LDS_LD + cch * 8]), g);
    }
  };

  const v8f zero = {0.f, 0.f, 0.f, 0.f, 0.f, 0.f, 0.f, 0.f};
  v8f acc[4];
#pragma unroll
  for (int j = 0; j < 4; ++j) acc[j] = zero;

  stage(0, 0);

  const int nIter = KIN / BLK_K;   // 32
#pragma unroll 1
  for (int kk = 0; kk < nIter; ++kk) {
    const int cur = kk & 1;
    wait_async0();        // this wave's DMA into buf[cur] complete
    __syncthreads();      // everyone's DMA complete; everyone done reading buf[cur^1]
    if (kk + 1 < nIter) stage(cur ^ 1, (kk + 1) * BLK_K);  // overlap with compute

    // Issue ALL fragment loads first (one ds clause), then the 4 WMMAs, so the
    // scheduler can use partial s_wait_dscnt instead of a full stall per WMMA.
    v16bf a = load_fragA(&sA[cur][0], wv * 16, LDS_LD, 0);
    v16bf bfrag[4];
#pragma unroll
    for (int j = 0; j < 4; ++j)
      bfrag[j] = load_fragB(&sB[cur][0], j * 16, LDS_LD, 0);
#pragma unroll
    for (int j = 0; j < 4; ++j)
      acc[j] = __builtin_amdgcn_wmma_f32_16x16x32_bf16(false, a, false, bfrag[j],
                                                       (short)0, acc[j], false, false);
  }

  const int n = laneid() & 15, hb = laneid() >> 4;
  const int row0 = rowG + wv * 16;
#pragma unroll
  for (int j = 0; j < 4; ++j) {
    const int   col  = n0 + (j << 4) + n;
    const float bias = bi[col];
#pragma unroll
    for (int r = 0; r < 8; ++r) {
      const int m = r + (hb << 3);
      pi[(size_t)(row0 + m) * SIXH + col] = acc[j][r] + bias;
    }
  }
}

// --------------------------------------------- fused recurrent step (one per t)
// Each wave: 16 batch rows x 16 h-columns; accumulates all 5 gate projections
// ps_g = h_in @ Ws[g*H:(g+1)*H]^T, then full gate math + state update.
__global__ void __launch_bounds__(256) lstm_step_kernel(const bf16* __restrict__ h_in,
                                                        bf16* __restrict__ h_out,
                                                        float* __restrict__ c,
                                                        const bf16* __restrict__ wsb,
                                                        const float* __restrict__ bs,
                                                        const float* __restrict__ pi,
                                                        const float* __restrict__ mask,
                                                        int t) {
  const int wave = blockIdx.x * (blockDim.x >> 5) + (threadIdx.x >> 5);
  const int nT   = wave & 31;   // 32 column tiles (H/16)
  const int mT   = wave >> 5;   // 16 row tiles (B/16)
  const int row0 = mT << 4;
  const int col0 = nT << 4;

  const v8f zero = {0.f, 0.f, 0.f, 0.f, 0.f, 0.f, 0.f, 0.f};
  v8f acc[5];
#pragma unroll
  for (int g = 0; g < 5; ++g) acc[g] = zero;

#pragma unroll 1
  for (int k = 0; k < H; k += 32) {
    // Load phase: A + all 5 gate B-fragments (one vmem clause), then 5 WMMAs
    // sharing the same A operand back-to-back.
    v16bf a = load_fragA(h_in, row0, H, k);
    v16bf bfrag[5];
#pragma unroll
    for (int g = 0; g < 5; ++g)
      bfrag[g] = load_fragB(wsb, g * H + col0, H, k);
#pragma unroll
    for (int g = 0; g < 5; ++g)
      acc[g] = __builtin_amdgcn_wmma_f32_16x16x32_bf16(false, a, false, bfrag[g],
                                                       (short)0, acc[g], false, false);
  }

  const int n = laneid() & 15, hb = laneid() >> 4;
  const int hcol = col0 + n;
  float bsv[5];
#pragma unroll
  for (int g = 0; g < 5; ++g) bsv[g] = bs[g * H + hcol];

  const float* piBase = pi + (size_t)t * B_B * SIXH;
#pragma unroll
  for (int r = 0; r < 8; ++r) {
    const int b = row0 + r + (hb << 3);            // batch index
    const float* prow = piBase + (size_t)b * SIXH;
    const float p0 = prow[0 * H + hcol];
    const float p1 = prow[1 * H + hcol];
    const float p2 = prow[2 * H + hcol];
    const float p3 = prow[3 * H + hcol];
    const float p4 = prow[4 * H + hcol];
    const float p5 = prow[5 * H + hcol];

    const float ig = sigm(p0 + acc[0][r] + bsv[0]);
    const float fg = sigm(p1 + acc[1][r] + bsv[1]);
    const float mi = tanhf(p2 + acc[2][r] + bsv[2]);
    const float og = sigm(p3 + acc[3][r] + bsv[3]);
    const float hg = sigm(p4 + acc[4][r] + bsv[4]);

    const size_t sidx = (size_t)b * H + hcol;
    const float cn = ig * mi + fg * c[sidx];
    float o = og * tanhf(cn);
    o = hg * o + (1.0f - hg) * p5;
    o *= mask[sidx];
    c[sidx]     = cn;
    h_out[sidx] = (bf16)o;
  }
}

// ------------------------------------------------------ output head (one per t)
__global__ void __launch_bounds__(256) pred_kernel(const bf16* __restrict__ h,
                                                   const bf16* __restrict__ wo,
                                                   const float* __restrict__ bo,
                                                   float* __restrict__ out) {
  const int wave = blockIdx.x * (blockDim.x >> 5) + (threadIdx.x >> 5);
  if (wave >= (B_B / 16) * (NCPAD / 16)) return;
  const int nT   = wave % (NCPAD / 16);
  const int mT   = wave / (NCPAD / 16);
  const int row0 = mT << 4;
  const int n0   = nT << 4;

  v8f acc = {0.f, 0.f, 0.f, 0.f, 0.f, 0.f, 0.f, 0.f};
#pragma unroll 1
  for (int k = 0; k < H; k += 32) {
    v16bf a = load_fragA(h, row0, H, k);
    v16bf b = load_fragB(wo, n0, H, k);
    acc = __builtin_amdgcn_wmma_f32_16x16x32_bf16(false, a, false, b,
                                                  (short)0, acc, false, false);
  }
  const int n = laneid() & 15, hb = laneid() >> 4;
  const int col = n0 + n;
  if (col < NC) {
    const float bias = bo[col];
#pragma unroll
    for (int r = 0; r < 8; ++r) {
      const int row = row0 + r + (hb << 3);
      out[(size_t)row * NC + col] = acc[r] + bias;
    }
  }
}

// ------------------------------------------------------------------------ host
extern "C" void kernel_launch(void* const* d_in, const int* in_sizes, int n_in,
                              void* d_out, int out_size, void* d_ws, size_t ws_size,
                              hipStream_t stream) {
  (void)in_sizes; (void)n_in; (void)out_size; (void)ws_size;
  const float* x      = (const float*)d_in[0];
  const int*   labels = (const int*)  d_in[1];
  const float* embed  = (const float*)d_in[2];
  const float* Wi     = (const float*)d_in[3];
  const float* bi     = (const float*)d_in[4];
  const float* Ws     = (const float*)d_in[5];
  const float* bs     = (const float*)d_in[6];
  const float* Wo     = (const float*)d_in[7];
  const float* bo     = (const float*)d_in[8];
  const float* mask   = (const float*)d_in[9];
  float* out = (float*)d_out;

  // Workspace carve-up (256-byte aligned slabs).
  uintptr_t cur = (uintptr_t)d_ws;
  auto take = [&](size_t bytes) -> void* {
    void* p = (void*)cur;
    cur += (bytes + 255) & ~(size_t)255;
    return p;
  };
  bf16*  tiB = (bf16*) take((size_t)T_T * B_B * KIN * sizeof(bf16)); // 33.5 MB
  bf16*  wiB = (bf16*) take((size_t)SIXH * KIN * sizeof(bf16));      //  6.3 MB
  bf16*  wsB = (bf16*) take((size_t)FIVEH * H * sizeof(bf16));       //  2.6 MB
  bf16*  woB = (bf16*) take((size_t)NCPAD * H * sizeof(bf16));       //  0.2 MB
  float* piF = (float*)take((size_t)T_T * B_B * SIXH * sizeof(float)); // 201 MB
  bf16*  h0  = (bf16*) take((size_t)B_B * H * sizeof(bf16));
  bf16*  h1  = (bf16*) take((size_t)B_B * H * sizeof(bf16));
  float* cSt = (float*)take((size_t)B_B * H * sizeof(float));

  // 1) weight conversion + input staging
  cvt_f32_bf16<<<(SIXH * KIN + 255) / 256, 256, 0, stream>>>(Wi, wiB, SIXH * KIN);
  cvt_f32_bf16<<<(FIVEH * H + 255) / 256, 256, 0, stream>>>(Ws, wsB, FIVEH * H);
  cvt_wo_kernel<<<(NCPAD * H + 255) / 256, 256, 0, stream>>>(Wo, woB);
  build_ti_kernel<<<(T_T * B_B * KIN) / 256, 256, 0, stream>>>(x, labels, embed, tiB);
  init_state_kernel<<<(B_B * H + 255) / 256, 256, 0, stream>>>(h0, cSt);

  // 2) hoisted input projection for all timesteps at once
  //    blocks = (T*B/128) * (6H/64) = 128 * 48 = 6144
  gemm_pi_kernel<<<(T_T * B_B / BLK_M) * (SIXH / BLK_N), 256, 0, stream>>>(tiB, wiB, bi, piF);

  // 3) sequential recurrence + output head
  bf16* hb[2] = { h0, h1 };
  for (int t = 0; t < T_T; ++t) {
    lstm_step_kernel<<<(16 * 32) / 8, 256, 0, stream>>>(
        hb[t & 1], hb[(t + 1) & 1], cSt, wsB, bs, piF, mask, t);
    pred_kernel<<<((B_B / 16) * (NCPAD / 16) + 7) / 8, 256, 0, stream>>>(
        hb[(t + 1) & 1], woB, bo, out + (size_t)t * B_B * NC);
  }
}
